// SpikingWANN_57604101374650
// MI455X (gfx1250) — 8-bit matrix operand order repaired and checked by reading, compiled, not run on hardware
//
#include <hip/hip_runtime.h>
#include <stdint.h>

typedef __attribute__((ext_vector_type(16))) int   v16i;
typedef __attribute__((ext_vector_type(8)))  float v8f;

#define N_IN    1024
#define N_H1    2048
#define N_H2    2048
#define N_OUT   10
#define BATCH   1024
#define T_STEPS 8
#define FAN_IN  32
#define BETA_F  0.9f
#define INV_TAU 0.1f
#define FP8_ONE 0x38u

__device__ __forceinline__ uint32_t permIdx(uint32_t i) {
  uint32_t lo = i & 7u;
  uint32_t b3 = (i >> 3) & 1u;
  uint32_t b4 = (i >> 4) & 1u;
  return (i & ~31u) | (b3 << 4) | (b4 << 3) | lo;
}

__global__ void zero_f32_kernel(float* __restrict__ p, int n) {
  int i = blockIdx.x * blockDim.x + threadIdx.x;
  if (i < n) p[i] = 0.0f;
}

__device__ __forceinline__ uint8_t f32_to_e4m3(float f) {
  if (f == 0.0f) return 0;
  uint32_t u = __float_as_uint(f);
  uint32_t sgn = (u >> 31) << 7;
  int e = (int)((u >> 23) & 0xFF) - 127;
  uint32_t m = (u >> 20) & 7;
  int eb = e + 7;
  if (eb <= 0) return (uint8_t)sgn;
  if (eb > 15) { eb = 15; m = 6; }
  return (uint8_t)(sgn | ((uint32_t)eb << 3) | m);
}

__global__ void build_dense_w_kernel(const int* __restrict__ src,
                                     const float* __restrict__ w,
                                     uint8_t* __restrict__ Wd,
                                     int nSrc) {
  __shared__ float row[2048];
  int d = blockIdx.x;
  for (int i = threadIdx.x; i < nSrc; i += blockDim.x) row[i] = 0.0f;
  __syncthreads();
  if (threadIdx.x == 0) {
    #pragma unroll
    for (int e = 0; e < FAN_IN; ++e) row[src[d * FAN_IN + e]] += w[d * FAN_IN + e];
  }
  __syncthreads();
  uint8_t* out = Wd + (size_t)d * nSrc;
  for (int i = threadIdx.x; i < nSrc; i += blockDim.x) out[i] = f32_to_e4m3(row[i]);
}

__device__ __forceinline__ void threefry2x32_42(uint32_t x0, uint32_t x1,
                                                uint32_t& o0, uint32_t& o1) {
  const uint32_t k0 = 0u, k1 = 42u;
  const uint32_t k2 = k0 ^ k1 ^ 0x1BD11BDAu;
  x0 += k0; x1 += k1;
#define TFR(r) { x0 += x1; x1 = (x1 << (r)) | (x1 >> (32 - (r))); x1 ^= x0; }
  TFR(13) TFR(15) TFR(26) TFR(6)
  x0 += k1; x1 += k2 + 1u;
  TFR(17) TFR(29) TFR(16) TFR(24)
  x0 += k2; x1 += k0 + 2u;
  TFR(13) TFR(15) TFR(26) TFR(6)
  x0 += k0; x1 += k1 + 3u;
  TFR(17) TFR(29) TFR(16) TFR(24)
  x0 += k1; x1 += k2 + 4u;
  TFR(13) TFR(15) TFR(26) TFR(6)
  x0 += k2; x1 += k0 + 5u;
#undef TFR
  o0 = x0; o1 = x1;
}

__global__ void encode_kernel(const float* __restrict__ x,
                              uint8_t* __restrict__ enc) {
  const int total = T_STEPS * BATCH * N_IN;
  const int half  = total / 2;
  int i = blockIdx.x * blockDim.x + threadIdx.x;
  if (i >= half) return;
  uint32_t b0, b1;
  threefry2x32_42((uint32_t)i, (uint32_t)(i + half), b0, b1);
  float u0 = __uint_as_float((b0 >> 9) | 0x3f800000u) - 1.0f;
  float u1 = __uint_as_float((b1 >> 9) | 0x3f800000u) - 1.0f;
  int r0 = i % (BATCH * N_IN);
  int r1 = (i + half) % (BATCH * N_IN);
  enc[permIdx((uint32_t)i)]          = (u0 < x[r0]) ? (uint8_t)FP8_ONE : (uint8_t)0;
  enc[permIdx((uint32_t)(i + half))] = (u1 < x[r1]) ? (uint8_t)FP8_ONE : (uint8_t)0;
}

__global__ __launch_bounds__(128)
void gemm_lif_kernel(const uint8_t* __restrict__ A,
                     const uint8_t* __restrict__ W,
                     float* __restrict__ v,
                     uint8_t* __restrict__ sOut,
                     int M, int K, int N) {
  const int tid   = threadIdx.x;
  const int lane  = tid & 31;
  const int wave  = tid >> 5;
  const int mBase = blockIdx.y * 32;
  const int nBase = blockIdx.x * 128 + wave * 32;
  const int ln15  = lane & 15;
  const int hi    = lane >> 4;

  union Frag { v16i v; uint4 q[4]; };
  struct Frags { Frag a[2]; Frag b[2]; };

  v8f acc[2][2] = {{{}, {}}, {{}, {}}};

  const uint8_t* aPtr = A + (size_t)(mBase + ln15) * K + hi * 16;
  const uint8_t* wPtr = W + (size_t)(nBase + ln15) * K + hi * 8;

  auto LOAD = [&](Frags& f, int k0) {
    #pragma unroll
    for (int mi = 0; mi < 2; ++mi) {
      const uint8_t* ap = aPtr + (size_t)mi * 16 * K + k0;
      #pragma unroll
      for (int q = 0; q < 4; ++q)
        f.a[mi].q[q] = *reinterpret_cast<const uint4*>(ap + 32 * q);
    }
    #pragma unroll
    for (int nj = 0; nj < 2; ++nj) {
      const uint8_t* wp = wPtr + (size_t)nj * 16 * K + k0;
      #pragma unroll
      for (int q = 0; q < 4; ++q)
      {
        uint2* d2 = reinterpret_cast<uint2*>(&f.b[nj].q[q]);
        d2[0] = *reinterpret_cast<const uint2*>(wp + 32 * q);
        d2[1] = *reinterpret_cast<const uint2*>(wp + 32 * q + 16);
      }
    }
  };
  auto COMP = [&](Frags& f) {
    #pragma unroll
    for (int mi = 0; mi < 2; ++mi)
      #pragma unroll
      for (int nj = 0; nj < 2; ++nj)
        acc[mi][nj] = __builtin_amdgcn_wmma_f32_16x16x128_fp8_fp8(
            f.a[mi].v, f.b[nj].v, (short)0, acc[mi][nj], false, false);
  };

  Frags f0, f1;
  LOAD(f0, 0);
  int k0 = 0;
  for (; k0 < K - 256; k0 += 256) {
    LOAD(f1, k0 + 128);
    COMP(f0);
    LOAD(f0, k0 + 256);
    COMP(f1);
  }
  LOAD(f1, k0 + 128);
  COMP(f0);
  COMP(f1);

  const int colPerm = ((ln15 & 8) << 1) + (ln15 & 7);
  #pragma unroll
  for (int mi = 0; mi < 2; ++mi) {
    #pragma unroll
    for (int nj = 0; nj < 2; ++nj) {
      #pragma unroll
      for (int r = 0; r < 8; ++r) {
        int mloc = mi * 16 + r + hi * 8;
        size_t idx = (size_t)(mBase + mloc) * N + (size_t)(nBase + colPerm + nj * 8);
        float vv = v[idx];
        vv = BETA_F * vv + INV_TAU * acc[mi][nj][r];
        bool s = (vv >= 1.0f);
        v[idx] = s ? 0.0f : vv;
        sOut[idx] = s ? (uint8_t)FP8_ONE : (uint8_t)0;
      }
    }
  }
}

__global__ void out_layer_kernel(const uint8_t* __restrict__ s2,
                                 const float* __restrict__ w2o,
                                 float* __restrict__ vo,
                                 float* __restrict__ out) {
  int i = blockIdx.x * blockDim.x + threadIdx.x;
  if (i >= BATCH * N_OUT) return;
  int b = i / N_OUT;
  int o = i % N_OUT;
  const uint8_t* sp = s2 + (size_t)b * N_H2;
  const float* wr = w2o + (size_t)o * N_H2;
  float acc = 0.0f;
  for (int k = 0; k < N_H2; k += 4) {
    uint32_t pk = *reinterpret_cast<const uint32_t*>(sp + k);
    int kk = (int)permIdx((uint32_t)k);
    acc += (pk & 0x000000FFu) ? wr[kk]     : 0.0f;
    acc += (pk & 0x0000FF00u) ? wr[kk + 1] : 0.0f;
    acc += (pk & 0x00FF0000u) ? wr[kk + 2] : 0.0f;
    acc += (pk & 0xFF000000u) ? wr[kk + 3] : 0.0f;
  }
  float vv = vo[i];
  vv = BETA_F * vv + INV_TAU * acc;
  float s = (vv >= 1.0f) ? 1.0f : 0.0f;
  vo[i] = (1.0f - s) * vv;
  out[i] += s;
}

extern "C" void kernel_launch(void* const* d_in, const int* in_sizes, int n_in,
                              void* d_out, int out_size, void* d_ws, size_t ws_size,
                              hipStream_t stream) {
  (void)in_sizes; (void)n_in; (void)out_size; (void)ws_size;

  const float* x     = (const float*)d_in[0];
  const float* w01   = (const float*)d_in[1];
  const float* w12   = (const float*)d_in[2];
  const float* w2o   = (const float*)d_in[3];
  const int*   src01 = (const int*)d_in[4];
  const int*   src12 = (const int*)d_in[6];
  float* out = (float*)d_out;

  char* ws = (char*)d_ws;
  size_t off = 0;
  uint8_t* W01d = (uint8_t*)(ws + off); off += (size_t)N_H1 * N_IN;
  uint8_t* W12d = (uint8_t*)(ws + off); off += (size_t)N_H2 * N_H1;
  uint8_t* enc  = (uint8_t*)(ws + off); off += (size_t)T_STEPS * BATCH * N_IN;
  uint8_t* s1   = (uint8_t*)(ws + off); off += (size_t)BATCH * N_H1;
  uint8_t* s2   = (uint8_t*)(ws + off); off += (size_t)BATCH * N_H2;
  float* v1 = (float*)(ws + off); off += (size_t)BATCH * N_H1 * 4;
  float* v2 = (float*)(ws + off); off += (size_t)BATCH * N_H2 * 4;
  float* vo = (float*)(ws + off); off += (size_t)BATCH * N_OUT * 4;

  zero_f32_kernel<<<(BATCH * N_H1 + 255) / 256, 256, 0, stream>>>(v1, BATCH * N_H1);
  zero_f32_kernel<<<(BATCH * N_H2 + 255) / 256, 256, 0, stream>>>(v2, BATCH * N_H2);
  zero_f32_kernel<<<(BATCH * N_OUT + 255) / 256, 256, 0, stream>>>(vo, BATCH * N_OUT);
  zero_f32_kernel<<<(BATCH * N_OUT + 255) / 256, 256, 0, stream>>>(out, BATCH * N_OUT);

  build_dense_w_kernel<<<N_H1, 256, 0, stream>>>(src01, w01, W01d, N_IN);
  build_dense_w_kernel<<<N_H2, 256, 0, stream>>>(src12, w12, W12d, N_H1);

  const int half = T_STEPS * BATCH * N_IN / 2;
  encode_kernel<<<(half + 255) / 256, 256, 0, stream>>>(x, enc);

  for (int t = 0; t < T_STEPS; ++t) {
    const uint8_t* encT = enc + (size_t)t * BATCH * N_IN;
    gemm_lif_kernel<<<dim3(N_H1 / 128, BATCH / 32), 128, 0, stream>>>(
        encT, W01d, v1, s1, BATCH, N_IN, N_H1);
    gemm_lif_kernel<<<dim3(N_H2 / 128, BATCH / 32), 128, 0, stream>>>(
        s1, W12d, v2, s2, BATCH, N_H1, N_H2);
    out_layer_kernel<<<(BATCH * N_OUT + 255) / 256, 256, 0, stream>>>(
        s2, w2o, vo, out);
  }
}
